// DecodePredictions_20641612824903
// MI455X (gfx1250) — compile-verified, matrix operand order checked by reading, not run on hardware
//
#include <hip/hip_runtime.h>

// Problem constants (from reference)
#define B_   16
#define N_   21824
#define C_   80
#define K_   512      // PRE_NMS_K
#define M_   100      // MAX_PER_CLASS == MAX_DET
#define CONF_THR_ 0.05f
#define IOU_THR_  0.5f

typedef __attribute__((ext_vector_type(16))) _Float16 v16h;
typedef __attribute__((ext_vector_type(8)))  float    v8f;

#if defined(__has_builtin)
#if __has_builtin(__builtin_amdgcn_global_load_async_to_lds_b128)
#define HAVE_ASYNC_LDS 1
#endif
#endif

#ifdef HAVE_ASYNC_LDS
// Exact parameter types per hipcc diagnostic: v4i pointer in AS1 (global) and
// the LDS destination in AS3.
typedef int v4i_vs __attribute__((vector_size(16)));
typedef __attribute__((address_space(1))) v4i_vs* g_v4i_p;
typedef __attribute__((address_space(3))) v4i_vs* l_v4i_p;
#endif

// ---------------------------------------------------------------------------
// Stage 1: decode boxes.  boxes[b][n] = {y1,x1,y2,x2} * stride
// Grid coords per reference's meshgrid('xy') + reshape: gy = local % fh,
// gx = local / fh   (levels: 128^2, 64^2, 32^2, 16^2, 8^2).
// ---------------------------------------------------------------------------
__global__ __launch_bounds__(256)
void decode_kernel(const float* __restrict__ regs, float* __restrict__ boxes) {
  int gid = blockIdx.x * blockDim.x + threadIdx.x;   // over B*N
  if (gid >= B_ * N_) return;
  int n = gid % N_;
  int local, fh; float s;
  if (n < 16384)      { local = n;          fh = 128; s = 8.0f;   }
  else if (n < 20480) { local = n - 16384;  fh = 64;  s = 16.0f;  }
  else if (n < 21504) { local = n - 20480;  fh = 32;  s = 32.0f;  }
  else if (n < 21760) { local = n - 21504;  fh = 16;  s = 64.0f;  }
  else                { local = n - 21760;  fh = 8;   s = 128.0f; }
  float gy = (float)(local % fh);
  float gx = (float)(local / fh);
  const float4 r = ((const float4*)regs)[gid];
  float x1 = (gx - r.x) * s;
  float x2 = (gx + r.y) * s;
  float y1 = (gy - r.z) * s;
  float y2 = (gy + r.w) * s;
  float4 bx; bx.x = y1; bx.y = x1; bx.z = y2; bx.w = x2;
  ((float4*)boxes)[gid] = bx;
}

// ---------------------------------------------------------------------------
// Stage 2: per-(b,c) top-512 radix select -> bitonic sort -> greedy NMS ->
// WMMA segmented-reduction ranks -> first-100-kept written to cs/cbx.
// One 512-thread (16 wave32) workgroup per (b,c).
// ---------------------------------------------------------------------------
__global__ __launch_bounds__(512)
void nms_kernel(const float* __restrict__ ctrs, const float* __restrict__ clfs,
                const float* __restrict__ boxes,
                float* __restrict__ cs, float* __restrict__ cbx) {
  __shared__ int hist[256];
  __shared__ unsigned long long skey[K_];
  __shared__ float4 cb[K_];
  __shared__ int keepf[K_];
  __shared__ float segsum[16];
  __shared__ int sT1, sCntGt, sThr16, sNeeded, sCnt, sCntEq;

  const int t  = threadIdx.x;
  const int bc = blockIdx.x;
  const int b  = bc / C_;
  const int c  = bc % C_;

  const float* ctr = ctrs + (size_t)b * N_;
  const float* clf = clfs + (size_t)b * N_ * C_ + c;

  float* csO = cs  + (size_t)bc * M_;
  float* cbO = cbx + (size_t)bc * M_ * 4;
  if (t < M_)     csO[t] = 0.0f;            // zero-pad per-class slots
  if (t < M_ * 4) cbO[t] = 0.0f;

  if (t < 256) hist[t] = 0;
  __syncthreads();

  // ---- pass 1: histogram of top byte of score bits (scores >= 0) ----
  for (int n = t; n < N_; n += K_) {
    if (n + K_ < N_) __builtin_prefetch(&clf[(size_t)(n + K_) * C_], 0, 0);
    float sc = clf[(size_t)n * C_] * ctr[n];
    atomicAdd(&hist[__float_as_uint(sc) >> 24], 1);
  }
  __syncthreads();
  if (t == 0) {
    int acc = 0, T1 = 0, cg = 0;
    for (int v = 255; v >= 0; --v) {
      acc += hist[v];
      if (acc >= K_) { T1 = v; cg = acc - hist[v]; break; }
    }
    sT1 = T1; sCntGt = cg;
  }
  __syncthreads();
  const int T1 = sT1, cntGt = sCntGt;
  if (t < 256) hist[t] = 0;
  __syncthreads();

  // ---- pass 2: second byte among elements whose top byte == T1 ----
  for (int n = t; n < N_; n += K_) {
    float sc = clf[(size_t)n * C_] * ctr[n];
    unsigned bits = __float_as_uint(sc);
    if ((int)(bits >> 24) == T1) atomicAdd(&hist[(bits >> 16) & 0xFF], 1);
  }
  __syncthreads();
  if (t == 0) {
    int acc = cntGt, T2 = 0, cg2 = cntGt;
    for (int v = 255; v >= 0; --v) {
      acc += hist[v];
      if (acc >= K_) { T2 = v; cg2 = acc - hist[v]; break; }
    }
    sThr16 = (T1 << 8) | T2;
    sNeeded = K_ - cg2;     // 1..hist[T2]
    sCnt = 0; sCntEq = 0;
  }
  __syncthreads();
  const int thr16 = sThr16, needed = sNeeded;

  // ---- pass 3: compact exactly 512 candidate keys ----
  for (int n = t; n < N_; n += K_) {
    float sc = clf[(size_t)n * C_] * ctr[n];
    unsigned bits = __float_as_uint(sc);
    int top16 = (int)(bits >> 16);
    int pos = -1;
    if (top16 > thr16) {
      pos = atomicAdd(&sCnt, 1);
    } else if (top16 == thr16) {
      int e = atomicAdd(&sCntEq, 1);
      if (e < needed) pos = atomicAdd(&sCnt, 1);
    }
    if (pos >= 0)
      skey[pos] = ((unsigned long long)bits << 32) | (unsigned)(~(unsigned)n);
  }
  __syncthreads();

  // ---- bitonic sort 512 keys descending (score desc, idx asc on ties) ----
  for (int k = 2; k <= K_; k <<= 1) {
    for (int j = k >> 1; j > 0; j >>= 1) {
      __syncthreads();
      int ix = t ^ j;
      if (ix > t) {
        unsigned long long a = skey[t], bb = skey[ix];
        bool up = ((t & k) == 0);
        if (up ? (a < bb) : (a > bb)) { skey[t] = bb; skey[ix] = a; }
      }
    }
  }
  __syncthreads();

  // ---- stage candidate boxes into LDS (async path when available) ----
  unsigned long long kk = skey[t];
  int   idx = (int)(~(unsigned)(kk & 0xFFFFFFFFull));
  float sc  = __uint_as_float((unsigned)(kk >> 32));
  const float4* gb = (const float4*)boxes + ((size_t)b * N_ + idx);
#ifdef HAVE_ASYNC_LDS
  __builtin_amdgcn_global_load_async_to_lds_b128((g_v4i_p)gb,
                                                 (l_v4i_p)&cb[t], 0, 0);
  asm volatile("s_wait_asynccnt 0" ::: "memory");
#else
  cb[t] = *gb;
#endif
  keepf[t] = (sc > CONF_THR_) ? 1 : 0;
  __syncthreads();

  // ---- greedy NMS: 512 broadcast steps ----
  const float4 mybox = cb[t];
  const float myArea = (mybox.z - mybox.x) * (mybox.w - mybox.y);
  for (int i = 0; i < K_; ++i) {
    if (keepf[i]) {
      if (t > i && keepf[t]) {
        float4 bi = cb[i];
        float yy1 = fmaxf(bi.x, mybox.x);
        float xx1 = fmaxf(bi.y, mybox.y);
        float yy2 = fminf(bi.z, mybox.z);
        float xx2 = fminf(bi.w, mybox.w);
        float inter = fmaxf(yy2 - yy1, 0.0f) * fmaxf(xx2 - xx1, 0.0f);
        float ai = (bi.z - bi.x) * (bi.w - bi.y);
        float iou = inter / fmaxf(ai + myArea - inter, 1e-8f);
        if (iou > IOU_THR_) keepf[t] = 0;
      }
    }
    __syncthreads();
  }

  // ---- WMMA segmented reduction: 16 per-wave segment sums of keep flags ---
  // A = 16x32 f16 (row m = keep[m*32 .. m*32+31], CDNA5 A layout),
  // B = ones -> D[m][*] = segment sum, via v_wmma_f32_16x16x32_f16.
  const int lane = t & 31;
  const int wave = t >> 5;
  if (wave == 0) {
    v16h a, bm;
    int m  = lane & 15;
    int kb = (lane >= 16) ? 8 : 0;
#pragma unroll
    for (int e = 0; e < 16; ++e) {
      int kpos = e + ((e >= 8) ? 8 : 0) + kb;   // A-matrix 16-bit K layout
      a[e]  = (_Float16)(float)keepf[m * 32 + kpos];
      bm[e] = (_Float16)1.0f;
    }
    v8f cz = {};
    v8f dd = __builtin_amdgcn_wmma_f32_16x16x32_f16(
        false, a, false, bm, (short)0, cz, false, false);
    if (lane == 0) {
#pragma unroll
      for (int v = 0; v < 8; ++v) segsum[v] = dd[v];       // M=0..7
    } else if (lane == 16) {
#pragma unroll
      for (int v = 0; v < 8; ++v) segsum[8 + v] = dd[v];   // M=8..15
    }
  }
  __syncthreads();

  // exclusive rank = sum of earlier segments + intra-wave ballot prefix
  const int myKeep = keepf[t];
  unsigned long long bal = __ballot(myKeep);
  int rank = __popcll(bal & ((1ull << lane) - 1ull));
  for (int sgi = 0; sgi < wave; ++sgi) rank += (int)segsum[sgi];

  // candidates are score-sorted, so first 100 kept == per-class top-100
  if (myKeep && rank < M_) {
    csO[rank] = sc;
    ((float4*)cbO)[rank] = mybox;
  }
}

// ---------------------------------------------------------------------------
// Stage 3: per-batch combine: top-100 of 80*100 candidates via bitonic sort
// of 8192 packed keys in 64KB LDS.  Output layout: fb[16][100][4] | fl | fs.
// ---------------------------------------------------------------------------
__global__ __launch_bounds__(1024)
void final_kernel(const float* __restrict__ cs, const float* __restrict__ cbx,
                  float* __restrict__ out) {
  __shared__ unsigned long long skey[8192];
  const int t = threadIdx.x;
  const int b = blockIdx.x;
  const float* csB = cs + (size_t)b * C_ * M_;

  for (int i = t; i < 8192; i += 1024) {
    if (i < C_ * M_) {
      unsigned bits = __float_as_uint(csB[i]);
      skey[i] = ((unsigned long long)bits << 32) | (unsigned)(~(unsigned)i);
    } else {
      skey[i] = 0ull;
    }
  }
  for (int k = 2; k <= 8192; k <<= 1) {
    for (int j = k >> 1; j > 0; j >>= 1) {
      __syncthreads();
      for (int i = t; i < 8192; i += 1024) {
        int ix = i ^ j;
        if (ix > i) {
          unsigned long long a = skey[i], bb = skey[ix];
          bool up = ((i & k) == 0);
          if (up ? (a < bb) : (a > bb)) { skey[i] = bb; skey[ix] = a; }
        }
      }
    }
  }
  __syncthreads();

  if (t < M_) {
    unsigned long long kk = skey[t];
    float fsv = __uint_as_float((unsigned)(kk >> 32));
    int flat  = (int)(~(unsigned)(kk & 0xFFFFFFFFull));   // c*100 + m
    bool valid = fsv > 0.0f;

    float* fb  = out + ((size_t)b * M_ + t) * 4;
    float* fl  = out + (size_t)B_ * M_ * 4 + (size_t)b * M_ + t;          // 6400
    float* fsp = out + (size_t)B_ * M_ * 4 + (size_t)B_ * M_
                     + (size_t)b * M_ + t;                                // 8000
    if (valid) {
      float4 bx = ((const float4*)cbx)[(size_t)b * C_ * M_ + flat];
      fb[0] = bx.x; fb[1] = bx.y; fb[2] = bx.z; fb[3] = bx.w;
      *fl = (float)(flat / M_);
    } else {
      fb[0] = 0.0f; fb[1] = 0.0f; fb[2] = 0.0f; fb[3] = 0.0f;
      *fl = 0.0f;
    }
    *fsp = fsv;
  }
}

// ---------------------------------------------------------------------------
extern "C" void kernel_launch(void* const* d_in, const int* in_sizes, int n_in,
                              void* d_out, int out_size, void* d_ws, size_t ws_size,
                              hipStream_t stream) {
  (void)in_sizes; (void)n_in; (void)out_size; (void)ws_size;
  const float* regs = (const float*)d_in[0];   // [B,N,4]
  const float* ctrs = (const float*)d_in[1];   // [B,N,1]
  const float* clfs = (const float*)d_in[2];   // [B,N,C]
  float* out = (float*)d_out;

  float* wsf   = (float*)d_ws;
  float* boxes = wsf;                                  // B*N*4   floats
  float* cs    = boxes + (size_t)B_ * N_ * 4;          // B*C*M   floats
  float* cbx   = cs    + (size_t)B_ * C_ * M_;         // B*C*M*4 floats

  decode_kernel<<<(B_ * N_ + 255) / 256, 256, 0, stream>>>(regs, boxes);
  nms_kernel<<<B_ * C_, 512, 0, stream>>>(ctrs, clfs, boxes, cs, cbx);
  final_kernel<<<B_, 1024, 0, stream>>>(cs, cbx, out);
}
